// TransformerModel_37898791420011
// MI455X (gfx1250) — compile-verified, matrix operand order checked by reading, not run on hardware
//
#include <hip/hip_runtime.h>
#include <hip/hip_bf16.h>
#include <math.h>

// ---------------------------------------------------------------------------
// GPT-style transformer forward (B=2,S=1024,H=1024,L=6,NH=16,HD=64,FF=4096,
// V=32000) for MI455X / gfx1250.  All GEMMs run through a single wave32
// WMMA kernel (v_wmma_f32_16x16x32_f16) with double-buffered LDS tiling.
// B (weight) tiles are fetched by the Tensor Data Mover (tensor_load_to_lds,
// synced with s_wait_tensorcnt) when the toolchain exposes the builtin;
// A tiles are register-staged and double-buffered so global latency hides
// under the WMMA pipe.
// ---------------------------------------------------------------------------

typedef _Float16 v16h __attribute__((ext_vector_type(16)));
typedef _Float16 v8h  __attribute__((ext_vector_type(8)));
typedef float    v8f  __attribute__((ext_vector_type(8)));
typedef unsigned int u32x4 __attribute__((ext_vector_type(4)));
typedef int      i32x8 __attribute__((ext_vector_type(8)));
typedef int      i32x4 __attribute__((ext_vector_type(4)));

#define WAVE 32  // gfx1250 is wave32-only

#if __has_builtin(__builtin_amdgcn_tensor_load_to_lds)
#define USE_TDM 1
#if __has_include(<hip/amd_detail/amd_gfx1250_TDM.h>)
// amdgpu-toolchain (clang-23 / therock-10.0 headers): 6-arg form
#define TDM_LOAD(g0, g1, g2, g3) \
  __builtin_amdgcn_tensor_load_to_lds((g0), (g1), (g2), (g3), (i32x8){0,0,0,0,0,0,0,0}, 0)
#else
// ROCm 7.2 (clang-22): 5-arg form
#define TDM_LOAD(g0, g1, g2, g3) \
  __builtin_amdgcn_tensor_load_to_lds((g0), (g1), (g2), (g3), 0)
#endif
#else
#define USE_TDM 0
#endif

__device__ inline v8h zero_v8h() {
  v8h z;
#pragma unroll
  for (int i = 0; i < 8; ++i) z[i] = (_Float16)0;
  return z;
}

// ---------------------------------------------------------------------------
// Tiled WMMA GEMM:  C[z][M,N] = A[z][M,K] * B[z][N,K]^T  (+ bias[n])
// A, B f16 row-major, contiguous K.  C f32 or f16.  K multiple of 32.
// Block = 256 threads = 8 waves; block tile 128x128xK32, double-buffered;
// each wave computes a 32x64 region = 2x4 WMMA tiles.
// ---------------------------------------------------------------------------
__global__ __launch_bounds__(256)
void gemm_f16_wmma(const _Float16* __restrict__ A, long lda, long sAz,
                   const _Float16* __restrict__ B, long ldb, long sBz,
                   void* __restrict__ Cv, long ldc, long sCz, int c_f16,
                   const float* __restrict__ bias,
                   int M, int N, int K)
{
  // 80B row pitch: 16B aligned for b128, 20-bank skew.  TDM reproduces the
  // pitch via pad_interval=16 DWORDs / pad_amount=4 DWORDs.
  __shared__ __align__(16) _Float16 As[2][128][40];
  __shared__ __align__(16) _Float16 Bs[2][128][40];

  const int tid  = threadIdx.x;
  const int wave = tid >> 5;          // 8 waves (wave32)
  const int lane = tid & (WAVE - 1);
  const int wm = wave >> 1;           // 0..3
  const int wn = wave & 1;            // 0..1
  const int q  = lane & 15;
  const int hl = lane >> 4;           // half-wave id

  const long bm = (long)blockIdx.y * 128;
  const long bn = (long)blockIdx.x * 128;
  const long z  = blockIdx.z;
  A += z * sAz;
  B += z * sBz;

  // this thread's two A-tile chunks (128x32 halves = 512 v8h chunks / 256 thr)
  const int row0 = tid >> 2,          ch0 = (tid & 3) * 8;
  const int row1 = (tid + 256) >> 2,  ch1 = ((tid + 256) & 3) * 8;
  long am0 = bm + row0; if (am0 >= M) am0 = M - 1;
  long am1 = bm + row1; if (am1 >= M) am1 = M - 1;
  const long bn0 = bn + row0, bn1 = bn + row1;

#if USE_TDM
  // Tensor Data Mover: one lane issues a 2D tile DMA (tile 32 x 128 elems of
  // 2B) from B[bn..bn+127, k0..k0+31] into Bs[buf], padded to 40-half pitch.
  // OOB rows (>= tensor_dim1 = N-bn) return zero per the TDM spec.
  auto issue_tdm_b = [&](long k0, int buf) {
    if (tid == 0) {
      // LDS aperture maps addr[31:0] directly -> low 32 bits of generic ptr.
      unsigned ldsoff = (unsigned)(unsigned long long)(const void*)&Bs[buf][0][0];
      unsigned long long ga = (unsigned long long)(const void*)(B + bn * ldb + k0);
      u32x4 g0;
      g0.x = 1u;                                            // count=1, user D#
      g0.y = ldsoff;                                        // lds_addr [63:32]
      g0.z = (unsigned)(ga & 0xFFFFFFFFu);                  // global_addr lo
      g0.w = (unsigned)((ga >> 32) & 0x01FFFFFFu) | (2u << 30); // addr[56:32] | type=2
      unsigned td0 = (unsigned)(K - k0);                    // dim0 extent left
      unsigned td1 = (unsigned)(N - bn);                    // rows left (OOB->0)
      unsigned long long st0 = (unsigned long long)ldb;     // dim0 stride
      i32x8 g1;
      g1[0] = (int)((1u << 16)        // data_size = 2 bytes
                  | (1u << 20)        // pad_enable
                  | (3u << 22)        // pad_interval = 16 DWORDs (one row)
                  | (3u << 25));      // pad_amount   = 4 DWORDs (8 halves)
      g1[1] = (int)((td0 & 0xFFFFu) << 16);                 // tensor_dim0 lo16
      g1[2] = (int)((td0 >> 16) | ((td1 & 0xFFFFu) << 16)); // dim0 hi | dim1 lo
      g1[3] = (int)((td1 >> 16) | (32u << 16));             // dim1 hi | tile_dim0=32
      g1[4] = (int)128;                                     // tile_dim1=128, tile_dim2=0
      g1[5] = (int)(st0 & 0xFFFFFFFFull);                   // dim0_stride lo32
      g1[6] = (int)((st0 >> 32) & 0xFFFFull);               // dim0_stride hi16
      g1[7] = 0;
      i32x4 gz = {0, 0, 0, 0};
      TDM_LOAD(g0, g1, gz, gz);
    }
  };
#endif

  v8f acc[2][4];
#pragma unroll
  for (int i = 0; i < 2; ++i)
#pragma unroll
    for (int j = 0; j < 4; ++j)
#pragma unroll
      for (int r = 0; r < 8; ++r) acc[i][j][r] = 0.0f;

  // ---- prologue: stage tile k=0 into buffer 0 ----
  v8h a0, a1;
#if USE_TDM
  issue_tdm_b(0, 0);
#else
  v8h b0, b1;
  b0 = (bn0 < N) ? *(const v8h*)(B + bn0 * ldb + ch0) : zero_v8h();
  b1 = (bn1 < N) ? *(const v8h*)(B + bn1 * ldb + ch1) : zero_v8h();
#endif
  a0 = *(const v8h*)(A + am0 * lda + ch0);
  a1 = *(const v8h*)(A + am1 * lda + ch1);
  *(v8h*)(&As[0][row0][ch0]) = a0;
  *(v8h*)(&As[0][row1][ch1]) = a1;
#if USE_TDM
  __builtin_amdgcn_s_wait_tensorcnt(0);
#else
  *(v8h*)(&Bs[0][row0][ch0]) = b0;
  *(v8h*)(&Bs[0][row1][ch1]) = b1;
#endif
  __syncthreads();

  for (long k0 = 0; k0 < K; k0 += 32) {
    const int cur = (int)((k0 >> 5) & 1);
    const int nxt = cur ^ 1;
    const bool has_next = (k0 + 32) < K;

    // ---- issue next tile's data movement before computing this one ----
    if (has_next) {
#if USE_TDM
      issue_tdm_b(k0 + 32, nxt);
#else
      b0 = (bn0 < N) ? *(const v8h*)(B + bn0 * ldb + (k0 + 32) + ch0) : zero_v8h();
      b1 = (bn1 < N) ? *(const v8h*)(B + bn1 * ldb + (k0 + 32) + ch1) : zero_v8h();
#endif
      a0 = *(const v8h*)(A + am0 * lda + (k0 + 32) + ch0);
      a1 = *(const v8h*)(A + am1 * lda + (k0 + 32) + ch1);
    }

    // ---- LDS -> fragments (ISA 16-bit A layout): two b128 reads each ----
    const int ko1 = hl * 8;        // lanes 0-15: K 0-7 ; lanes 16-31: K 8-15
    const int ko2 = 16 + hl * 8;   // lanes 0-15: K 16-23; lanes 16-31: K 24-31
    v16h af[2];
#pragma unroll
    for (int i = 0; i < 2; ++i) {
      int r = wm * 32 + i * 16 + q;
      v8h lo = *(const v8h*)(&As[cur][r][ko1]);
      v8h hi = *(const v8h*)(&As[cur][r][ko2]);
      af[i] = __builtin_shufflevector(lo, hi, 0,1,2,3,4,5,6,7,8,9,10,11,12,13,14,15);
    }
    v16h bf[4];
#pragma unroll
    for (int j = 0; j < 4; ++j) {
      int r = wn * 64 + j * 16 + q;
      v8h lo = *(const v8h*)(&Bs[cur][r][ko1]);
      v8h hi = *(const v8h*)(&Bs[cur][r][ko2]);
      bf[j] = __builtin_shufflevector(lo, hi, 0,1,2,3,4,5,6,7,8,9,10,11,12,13,14,15);
    }

#pragma unroll
    for (int i = 0; i < 2; ++i)
#pragma unroll
      for (int j = 0; j < 4; ++j)
        acc[i][j] = __builtin_amdgcn_wmma_f32_16x16x32_f16(
            /*neg_a=*/false, af[i], /*neg_b=*/false, bf[j],
            /*c_mod=*/(short)0, acc[i][j],
            /*reuse_a=*/false, /*reuse_b=*/false);

    // ---- commit next tile to the other LDS buffer ----
    if (has_next) {
      *(v8h*)(&As[nxt][row0][ch0]) = a0;
      *(v8h*)(&As[nxt][row1][ch1]) = a1;
#if USE_TDM
      __builtin_amdgcn_s_wait_tensorcnt(0);   // wave 0 waits for DMA; others pass
#else
      *(v8h*)(&Bs[nxt][row0][ch0]) = b0;
      *(v8h*)(&Bs[nxt][row1][ch1]) = b1;
#endif
      __syncthreads();
    }
  }

  // ---- store C (f32 layout: VGPR r -> M = r + 8*half, N = lane%16) ----
  float*    Cf = (float*)Cv;
  _Float16* Ch = (_Float16*)Cv;
#pragma unroll
  for (int i = 0; i < 2; ++i)
#pragma unroll
    for (int j = 0; j < 4; ++j)
#pragma unroll
      for (int r = 0; r < 8; ++r) {
        long m = bm + wm * 32 + i * 16 + hl * 8 + r;
        long n = bn + wn * 64 + j * 16 + q;
        if (m < M && n < N) {
          float val = acc[i][j][r];
          if (bias) val += bias[n];
          long idx = z * sCz + m * ldc + n;
          if (c_f16) Ch[idx] = (_Float16)val;
          else       Cf[idx] = val;
        }
      }
}

// ---------------------------------------------------------------------------
// fp32 [K,N] -> f16 [N,K] transpose (weight prep)
// ---------------------------------------------------------------------------
__global__ void transpose_w_kernel(const float* __restrict__ in,
                                   _Float16* __restrict__ out,
                                   long Ncols, long Krows, long total)
{
  long i = (long)blockIdx.x * 256 + threadIdx.x;
  if (i >= total) return;
  long n = i / Krows;
  long k = i - n * Krows;
  out[i] = (_Float16)in[k * Ncols + n];
}

// ---------------------------------------------------------------------------
// embedding gather: x[t, c] = tok[ids[t], c] + pos[t % S, c]
// ---------------------------------------------------------------------------
__global__ void embed_kernel(const int* __restrict__ ids,
                             const float* __restrict__ tok,
                             const float* __restrict__ pos,
                             float* __restrict__ x, long total)
{
  long i = (long)blockIdx.x * 256 + threadIdx.x;
  if (i >= total) return;
  long t = i >> 10;          // H = 1024
  long c = i & 1023;
  long srow = t & 1023;      // S = 1024
  long id = (long)ids[t];
  x[i] = tok[id * 1024 + c] + pos[srow * 1024 + c];
}

// ---------------------------------------------------------------------------
// LayerNorm row kernel -> f16 output (one 256-thread block per row of 1024)
// ---------------------------------------------------------------------------
__global__ void ln_kernel(const float* __restrict__ x,
                          const float* __restrict__ s,
                          const float* __restrict__ b,
                          _Float16* __restrict__ out, int Hd)
{
  __shared__ float red[256];
  long row = blockIdx.x;
  const float* xr = x + row * Hd;
  float sum = 0.f, sq = 0.f;
  for (int c = threadIdx.x; c < Hd; c += 256) {
    float v = xr[c]; sum += v; sq += v * v;
  }
  red[threadIdx.x] = sum; __syncthreads();
  for (int o = 128; o > 0; o >>= 1) {
    if (threadIdx.x < o) red[threadIdx.x] += red[threadIdx.x + o];
    __syncthreads();
  }
  float mean = red[0] / Hd; __syncthreads();
  red[threadIdx.x] = sq; __syncthreads();
  for (int o = 128; o > 0; o >>= 1) {
    if (threadIdx.x < o) red[threadIdx.x] += red[threadIdx.x + o];
    __syncthreads();
  }
  float var = red[0] / Hd - mean * mean;
  float rs  = rsqrtf(var + 1e-5f);
  for (int c = threadIdx.x; c < Hd; c += 256)
    out[row * Hd + c] = (_Float16)((xr[c] - mean) * rs * s[c] + b[c]);
}

// ---------------------------------------------------------------------------
// softmax over rows of length 1024 (scale folded in) -> f16
// ---------------------------------------------------------------------------
__global__ void softmax_kernel(const float* __restrict__ in,
                               _Float16* __restrict__ out,
                               int n, float scale)
{
  __shared__ float red[256];
  long row = blockIdx.x;
  const float* r = in + row * (long)n;
  _Float16* o = out + row * (long)n;
  float mx = -3.0e38f;
  float vals[4];
#pragma unroll
  for (int j = 0; j < 4; ++j) {
    int c = threadIdx.x + j * 256;
    vals[j] = r[c] * scale;
    mx = fmaxf(mx, vals[j]);
  }
  red[threadIdx.x] = mx; __syncthreads();
  for (int of = 128; of > 0; of >>= 1) {
    if (threadIdx.x < of) red[threadIdx.x] = fmaxf(red[threadIdx.x], red[threadIdx.x + of]);
    __syncthreads();
  }
  float m = red[0]; __syncthreads();
  float sum = 0.f;
#pragma unroll
  for (int j = 0; j < 4; ++j) { vals[j] = expf(vals[j] - m); sum += vals[j]; }
  red[threadIdx.x] = sum; __syncthreads();
  for (int of = 128; of > 0; of >>= 1) {
    if (threadIdx.x < of) red[threadIdx.x] += red[threadIdx.x + of];
    __syncthreads();
  }
  float inv = 1.0f / red[0];
#pragma unroll
  for (int j = 0; j < 4; ++j) o[threadIdx.x + j * 256] = (_Float16)(vals[j] * inv);
}

// ---------------------------------------------------------------------------
// V transpose: vT[(b,h,d),s] = qkv_h[(b,s), 2H + h*64 + d]
// ---------------------------------------------------------------------------
__global__ void vtrans_kernel(const _Float16* __restrict__ qkv,
                              _Float16* __restrict__ vT, long total)
{
  long i = (long)blockIdx.x * 256 + threadIdx.x;
  if (i >= total) return;
  long s = i & 1023; long r = i >> 10;     // S = 1024
  long d = r & 63;   r >>= 6;              // HD = 64
  long h = r & 15;   long b = r >> 4;      // NH = 16
  vT[i] = qkv[(b * 1024 + s) * 3072 + 2048 + h * 64 + d];
}

// ---------------------------------------------------------------------------
// residual add: x += t
// ---------------------------------------------------------------------------
__global__ void add_kernel(float* __restrict__ x, const float* __restrict__ t, long total)
{
  long i = (long)blockIdx.x * 256 + threadIdx.x;
  if (i >= total) return;
  x[i] += t[i];
}

// ---------------------------------------------------------------------------
// exact GELU -> f16
// ---------------------------------------------------------------------------
__global__ void gelu_kernel(const float* __restrict__ in, _Float16* __restrict__ out, long total)
{
  long i = (long)blockIdx.x * 256 + threadIdx.x;
  if (i >= total) return;
  float v = in[i];
  out[i] = (_Float16)(0.5f * v * (1.0f + erff(v * 0.70710678118654752f)));
}

// ---------------------------------------------------------------------------
// host side
// ---------------------------------------------------------------------------
static inline void run_gemm(const _Float16* A, long lda, long sAz,
                            const _Float16* Bt, long ldb, long sBz,
                            void* C, long ldc, long sCz, int cf16,
                            const float* bias, int M, int N, int K, int Z,
                            hipStream_t st)
{
  dim3 g((N + 127) / 128, (M + 127) / 128, Z);
  gemm_f16_wmma<<<g, 256, 0, st>>>(A, lda, sAz, Bt, ldb, sBz, C, ldc, sCz, cf16, bias, M, N, K);
}

extern "C" void kernel_launch(void* const* d_in, const int* in_sizes, int n_in,
                              void* d_out, int out_size, void* d_ws, size_t ws_size,
                              hipStream_t stream)
{
  (void)in_sizes; (void)n_in; (void)out_size; (void)ws_size;
  const long Vv = 32000, H = 1024, Ll = 6, NH = 16, HD = 64, FF = 4096;
  const long Bb = 2, S = 1024, M = Bb * S, H3 = 3 * H;

  const int*   ids      = (const int*)  d_in[0];
  const float* tok      = (const float*)d_in[1];
  const float* pos      = (const float*)d_in[2];
  const float* ln1_s    = (const float*)d_in[3];
  const float* ln1_b    = (const float*)d_in[4];
  const float* qkv_w    = (const float*)d_in[5];
  const float* qkv_b    = (const float*)d_in[6];
  const float* out_w    = (const float*)d_in[7];
  const float* out_b    = (const float*)d_in[8];
  const float* ln2_s    = (const float*)d_in[9];
  const float* ln2_b    = (const float*)d_in[10];
  const float* w1       = (const float*)d_in[11];
  const float* b1       = (const float*)d_in[12];
  const float* w2       = (const float*)d_in[13];
  const float* b2       = (const float*)d_in[14];
  const float* lnf_s    = (const float*)d_in[15];
  const float* lnf_b    = (const float*)d_in[16];
  const float* lm_w     = (const float*)d_in[17];
  float* logits = (float*)d_out;

  // workspace carve-up (256B aligned)
  size_t off = 0;
  auto alloc = [&](size_t bytes) -> void* {
    size_t cur = (off + 255) & ~(size_t)255;
    off = cur + bytes;
    return (void*)((char*)d_ws + cur);
  };
  _Float16* qkv_wT = (_Float16*)alloc(Ll * H3 * H * 2);
  _Float16* out_wT = (_Float16*)alloc(Ll * H * H * 2);
  _Float16* w1T    = (_Float16*)alloc(Ll * FF * H * 2);
  _Float16* w2T    = (_Float16*)alloc(Ll * H * FF * 2);
  _Float16* lm_wT  = (_Float16*)alloc(Vv * H * 2);
  float*    x      = (float*)   alloc(M * H * 4);
  _Float16* h16    = (_Float16*)alloc(M * H * 2);
  _Float16* qkv_h  = (_Float16*)alloc(M * H3 * 2);
  float*    scores = (float*)   alloc(Bb * NH * S * S * 4);
  _Float16* attn16 = (_Float16*)alloc(Bb * NH * S * S * 2);
  _Float16* vT     = (_Float16*)alloc(Bb * NH * HD * S * 2);
  _Float16* o16    = (_Float16*)alloc(M * H * 2);
  float*    tmp    = (float*)   alloc(M * FF * 4);
  _Float16* ff16   = (_Float16*)alloc(M * FF * 2);

  auto tw = [&](const float* in, _Float16* out, long Ncols, long Krows) {
    long total = Ncols * Krows;
    transpose_w_kernel<<<(total + 255) / 256, 256, 0, stream>>>(in, out, Ncols, Krows, total);
  };

  // ---- weight prep: fp32 [K,N] -> f16 [N,K] ----
  for (long l = 0; l < Ll; ++l) {
    tw(qkv_w + l * H * H3, qkv_wT + l * H3 * H, H3, H);
    tw(out_w + l * H * H,  out_wT + l * H * H,  H,  H);
    tw(w1    + l * H * FF, w1T    + l * FF * H, FF, H);
    tw(w2    + l * FF * H, w2T    + l * H * FF, H,  FF);
  }
  tw(lm_w, lm_wT, Vv, H);

  // ---- embeddings ----
  embed_kernel<<<(M * H + 255) / 256, 256, 0, stream>>>(ids, tok, pos, x, M * H);

  // ---- layers ----
  for (long l = 0; l < Ll; ++l) {
    // attention
    ln_kernel<<<M, 256, 0, stream>>>(x, ln1_s + l * H, ln1_b + l * H, h16, (int)H);
    run_gemm(h16, H, 0, qkv_wT + l * H3 * H, H, 0,
             qkv_h, H3, 0, 1, qkv_b + l * H3, (int)M, (int)H3, (int)H, 1, stream);
    vtrans_kernel<<<(Bb * NH * HD * S + 255) / 256, 256, 0, stream>>>(qkv_h, vT, Bb * NH * HD * S);
    for (long b = 0; b < Bb; ++b)   // scores[b,h] = q @ k^T, batched over heads
      run_gemm(qkv_h + b * S * H3,     H3, HD,
               qkv_h + b * S * H3 + H, H3, HD,
               scores + b * NH * S * S, S, S * S, 0, nullptr,
               (int)S, (int)S, (int)HD, (int)NH, stream);
    softmax_kernel<<<Bb * NH * S, 256, 0, stream>>>(scores, attn16, (int)S, 0.125f);
    for (long b = 0; b < Bb; ++b)   // o[b,h] = attn @ v, batched over heads
      run_gemm(attn16 + b * NH * S * S, S, S * S,
               vT + b * NH * HD * S,    S, HD * S,
               o16 + b * S * H, H, HD, 1, nullptr,
               (int)S, (int)HD, (int)S, (int)NH, stream);
    run_gemm(o16, H, 0, out_wT + l * H * H, H, 0,
             tmp, H, 0, 0, out_b + l * H, (int)M, (int)H, (int)H, 1, stream);
    add_kernel<<<(M * H + 255) / 256, 256, 0, stream>>>(x, tmp, M * H);

    // mlp
    ln_kernel<<<M, 256, 0, stream>>>(x, ln2_s + l * H, ln2_b + l * H, h16, (int)H);
    run_gemm(h16, H, 0, w1T + l * FF * H, H, 0,
             tmp, FF, 0, 0, b1 + l * FF, (int)M, (int)FF, (int)H, 1, stream);
    gelu_kernel<<<(M * FF + 255) / 256, 256, 0, stream>>>(tmp, ff16, M * FF);
    run_gemm(ff16, FF, 0, w2T + l * H * FF, FF, 0,
             tmp, H, 0, 0, b2 + l * H, (int)M, (int)H, (int)FF, 1, stream);
    add_kernel<<<(M * H + 255) / 256, 256, 0, stream>>>(x, tmp, M * H);
  }

  // ---- final LN + logits ----
  ln_kernel<<<M, 256, 0, stream>>>(x, lnf_s, lnf_b, h16, (int)H);
  run_gemm(h16, H, 0, lm_wT, H, 0,
           logits, Vv, 0, 0, nullptr, (int)M, (int)Vv, (int)H, 1, stream);
}
